// GroupedTernaryLinear_42039139893446
// MI455X (gfx1250) — matrix operand order repaired and checked by reading, not compiled, not run on hardware
//
#include <hip/hip_runtime.h>
#include <stdint.h>

// ---------------------------------------------------------------------------
// GroupedTernaryLinear for MI455X (gfx1250, wave32, WMMA)
//   out[b,t, g*1024+o] = sum_i rmsnorm(x)[b,t, g*1024+i] * Wq[g*1024+o, i]
//   Wq = ternary(bf16(weight)) with per-64-chunk scale; math in bf16 WMMA,
//   f32 accumulate (v_wmma_f32_16x16x32_bf16).
//   Wave tile: 32M x 64N  (8 WMMA per 12 b128 loads per K=32 step)
// ---------------------------------------------------------------------------

typedef __attribute__((ext_vector_type(16))) __bf16 bf16x16;
typedef __attribute__((ext_vector_type(8)))  float  f32x8;
typedef __attribute__((ext_vector_type(4)))  float  f32x4;   // native vec for NT loads

#define K_TOT  4096   // x row length
#define K_GRP  1024   // per-group inner dim
#define N_GRP  1024   // per-group output dim
#define N_TOT  4096
#define M_TOT  8192   // 2*4096 tokens

__device__ __forceinline__ unsigned short f2bf(float f) {
  uint32_t u = __float_as_uint(f);
  u += 0x7FFFu + ((u >> 16) & 1u);          // round-to-nearest-even
  return (unsigned short)(u >> 16);
}
__device__ __forceinline__ float bf2f(unsigned short b) {
  return __uint_as_float(((uint32_t)b) << 16);
}

// --------------------------- weight ternarization ---------------------------
// one wave32 per flat group of 64; lane handles 2 consecutive weights
__global__ __launch_bounds__(256) void k_quant_w(const float* __restrict__ w,
                                                 unsigned short* __restrict__ wq) {
  const int lane = threadIdx.x & 31;
  const int gid  = blockIdx.x * 8 + (threadIdx.x >> 5);   // 65536 groups total
  const float2 wv = ((const float2*)(w + (size_t)gid * 64))[lane];
  const float a = bf2f(f2bf(wv.x));                        // bf16 cast like torch .bfloat16()
  const float b = bf2f(f2bf(wv.y));
  float s = fabsf(a) + fabsf(b);
#pragma unroll
  for (int off = 16; off >= 1; off >>= 1) s += __shfl_xor(s, off, 32);
  const float scale = fmaxf(s * (1.0f / 64.0f), 1e-8f);
  const float inv   = 1.0f / scale;
  const float qa = fminf(fmaxf(rintf(a * inv), -1.f), 1.f) * scale;
  const float qb = fminf(fmaxf(rintf(b * inv), -1.f), 1.f) * scale;
  ushort2 o; o.x = f2bf(qa); o.y = f2bf(qb);
  ((ushort2*)(wq + (size_t)gid * 64))[lane] = o;
}

// --------------------------- rms-norm + bf16 cast ---------------------------
// one 256-thread block per row of 4096 f32
__global__ __launch_bounds__(256) void k_rmsnorm_cast(const float* __restrict__ x,
                                                      unsigned short* __restrict__ xn) {
  const int row = blockIdx.x;
  const int t   = threadIdx.x;
  const f32x4* xr = (const f32x4*)(x + (size_t)row * K_TOT);
  f32x4 v[4];
  float ss = 0.f;
#pragma unroll
  for (int i = 0; i < 4; ++i) {
    v[i] = __builtin_nontemporal_load(&xr[i * 256 + t]);   // x read exactly once
    ss += v[i].x * v[i].x + v[i].y * v[i].y + v[i].z * v[i].z + v[i].w * v[i].w;
  }
#pragma unroll
  for (int off = 16; off >= 1; off >>= 1) ss += __shfl_xor(ss, off, 32);
  __shared__ float part[8];
  __shared__ float stot;
  if ((t & 31) == 0) part[t >> 5] = ss;
  __syncthreads();
  if (t == 0) {
    float s = 0.f;
#pragma unroll
    for (int i = 0; i < 8; ++i) s += part[i];
    stot = s;
  }
  __syncthreads();
  const float scale = rsqrtf(stot * (1.0f / K_TOT) + 1.1920928955078125e-7f); // FLT_EPSILON
  ushort4* xo = (ushort4*)(xn + (size_t)row * K_TOT);
#pragma unroll
  for (int i = 0; i < 4; ++i) {
    ushort4 o;
    o.x = f2bf(v[i].x * scale);
    o.y = f2bf(v[i].y * scale);
    o.z = f2bf(v[i].z * scale);
    o.w = f2bf(v[i].w * scale);
    xo[i * 256 + t] = o;
  }
}

// ------------------------------- WMMA GEMM ----------------------------------
struct B32Pair { uint4 lo, hi; };

// A operand (16x32 bf16): lane holds row (lane&15), K chunks (lane>>4)*8+{0..7}
// and +16+{0..7}  (ISA 7.12.2 16-bit A table)
__device__ __forceinline__ bf16x16 load_a_tile(const unsigned short* p) {
  B32Pair t;
  t.lo = *(const uint4*)(p);
  t.hi = *(const uint4*)(p + 16);
  return __builtin_bit_cast(bf16x16, t);
}
// B operand (32x16 bf16): lane holds column (lane&15), K chunks (lane>>4)*8+{0..7}
// and +16+{0..7}  (same K order as A)
__device__ __forceinline__ bf16x16 load_b_tile(const unsigned short* p) {
  B32Pair t;
  t.lo = *(const uint4*)(p);
  t.hi = *(const uint4*)(p + 16);
  return __builtin_bit_cast(bf16x16, t);
}

__global__ __launch_bounds__(256) void k_gemm_wmma(const unsigned short* __restrict__ xn,
                                                   const unsigned short* __restrict__ wq,
                                                   float* __restrict__ out) {
  const int lane = threadIdx.x & 31;
  const int wave = threadIdx.x >> 5;
  const int g  = blockIdx.z;
  const int m0 = blockIdx.y * 32;                 // 32 rows per wave tile
  const int n0 = blockIdx.x * 512 + wave * 64;    // 64 cols per wave tile
  const int lm = lane & 15;
  const int hi = lane >> 4;

  const unsigned short* a0 = xn + (size_t)(m0 + lm) * K_TOT + g * K_GRP + hi * 8;
  const unsigned short* a1 = a0 + (size_t)16 * K_TOT;
  const unsigned short* bp = wq + (size_t)(g * N_GRP + n0 + lm) * K_GRP + hi * 8;

  f32x8 c[2][4] = {};

#pragma unroll 2
  for (int k = 0; k < K_GRP; k += 32) {
    const bf16x16 A0 = load_a_tile(a0 + k);
    const bf16x16 A1 = load_a_tile(a1 + k);
    bf16x16 B[4];
#pragma unroll
    for (int j = 0; j < 4; ++j) B[j] = load_b_tile(bp + (size_t)j * 16 * K_GRP + k);
#pragma unroll
    for (int j = 0; j < 4; ++j) {
      c[0][j] = __builtin_amdgcn_wmma_f32_16x16x32_bf16(false, A0, false, B[j], (short)0, c[0][j], false, false);
      c[1][j] = __builtin_amdgcn_wmma_f32_16x16x32_bf16(false, A1, false, B[j], (short)0, c[1][j], false, false);
    }
  }

  // D layout: VGPR v -> M = (lane>>4)*8 + v, N = lane&15; stream out with NT hint
  float* ob = out + (size_t)(m0 + hi * 8) * N_TOT + g * N_GRP + n0 + lm;
#pragma unroll
  for (int i = 0; i < 2; ++i)
#pragma unroll
    for (int j = 0; j < 4; ++j)
#pragma unroll
      for (int v = 0; v < 8; ++v)
        __builtin_nontemporal_store(c[i][j][v], ob + (size_t)(i * 16 + v) * N_TOT + j * 16);
}

// ------------------------------- launcher -----------------------------------
extern "C" void kernel_launch(void* const* d_in, const int* in_sizes, int n_in,
                              void* d_out, int out_size, void* d_ws, size_t ws_size,
                              hipStream_t stream) {
  (void)in_sizes; (void)n_in; (void)out_size; (void)ws_size;
  const float* x = (const float*)d_in[0];   // (2,4096,4096) f32
  const float* w = (const float*)d_in[1];   // (4096,1024)   f32
  float* out = (float*)d_out;               // (2,4096,4096) f32

  unsigned short* wq = (unsigned short*)d_ws;                              // 8 MiB
  unsigned short* xn = (unsigned short*)((char*)d_ws + ((size_t)8 << 20)); // 64 MiB

  k_quant_w<<<8192, 256, 0, stream>>>(w, wq);          // 65536 groups, wave/group
  k_rmsnorm_cast<<<M_TOT, 256, 0, stream>>>(x, xn);    // block/row
  dim3 grid(N_GRP / 512, M_TOT / 32, 4);               // (2, 256, 4), 8 waves/block
  k_gemm_wmma<<<grid, 256, 0, stream>>>(xn, wq, out);
}
